// Deepfm_2156073583145
// MI455X (gfx1250) — matrix operand order repaired and checked by reading, not compiled, not run on hardware
//
#include <hip/hip_runtime.h>
#include <hip/hip_bf16.h>
#include <math.h>

typedef __attribute__((ext_vector_type(16))) _Float16 v16h;
typedef __attribute__((ext_vector_type(8)))  _Float16 v8h;
typedef __attribute__((ext_vector_type(8)))  float    v8f;
typedef __attribute__((ext_vector_type(4)))  float    v4f;

#define BATCH 16384
#define NFEAT 26
#define NDENSE 13
#define VOCAB 100000
#define EMB 16
#define HIDDEN 400
#define DNN_IN 429        // 26*16 + 13
#define K0PAD 448         // 429 -> 14 k-tiles of 32
#define K1PAD 416         // 400 -> 13 k-tiles of 32
#define NTILES 25         // 400 / 16

// ---------------------------------------------------------------------------
// Pack fp32 weight matrix W[K x N] (row-major, stride N) into WMMA B-fragment
// order (f16): Wp[((kt*NT + nt)*32 + lane)*16 + i] = W[kt*32 + kmapB(i,lane)][nt*16 + (lane&15)]
// B-matrix 32x16 f16 layout: lane l holds column n = l&15; element i -> K = 16*(i>=8) + 8*(l>=16) + (i&7).
// ---------------------------------------------------------------------------
__global__ __launch_bounds__(256)
void pack_w_kernel(const float* __restrict__ W, _Float16* __restrict__ Wp,
                   int K, int KT, int N)
{
    int tid = blockIdx.x * 256 + threadIdx.x;
    int total = KT * (N / 16) * 512;
    if (tid >= total) return;
    int i    = tid & 15;
    int lane = (tid >> 4) & 31;
    int tile = tid >> 9;
    int NT   = N / 16;
    int kt   = tile / NT;
    int nt   = tile % NT;
    int k    = kt * 32 + ((i >> 3) << 4) + ((lane >> 4) << 3) + (i & 7);
    int n    = nt * 16 + (lane & 15);
    float v  = (k < K) ? W[(size_t)k * N + n] : 0.f;
    Wp[tid] = (_Float16)v;
}

// ---------------------------------------------------------------------------
// Fused embedding gather + FM (linear + 2nd-order cross) + DNN-input build.
// One wave per batch row. Lanes 0..25 own one sparse feature each.
// h0 layout: [B x 448] f16: cols 0..415 = fm (26x16), 416..428 = dense, 429..447 = 0
// logit[b]  = linear_sparse + linear_dense + 0.5*(sum(s^2) - sum(fm^2))
// ---------------------------------------------------------------------------
__global__ __launch_bounds__(256)
void fm_embed_kernel(const int* __restrict__ xs, const float* __restrict__ xd,
                     const float* __restrict__ emb1, const float* __restrict__ emb2,
                     const float* __restrict__ w_dense,
                     _Float16* __restrict__ h0, float* __restrict__ logit)
{
    const int lane = threadIdx.x & 31;
    const int b    = blockIdx.x * 8 + (threadIdx.x >> 5);

    float fm[16];
#pragma unroll
    for (int e = 0; e < 16; e++) fm[e] = 0.f;
    float lin = 0.f, sumsq = 0.f;

    if (lane < NFEAT) {
        int idx = xs[b * NFEAT + lane];
        const float* ep = emb2 + ((size_t)lane * VOCAB + (size_t)idx) * EMB;
        v4f e0 = *(const v4f*)(ep);
        v4f e1 = *(const v4f*)(ep + 4);
        v4f e2 = *(const v4f*)(ep + 8);
        v4f e3 = *(const v4f*)(ep + 12);
#pragma unroll
        for (int j = 0; j < 4; j++) { fm[j] = e0[j]; fm[4+j] = e1[j]; fm[8+j] = e2[j]; fm[12+j] = e3[j]; }
#pragma unroll
        for (int e = 0; e < 16; e++) sumsq += fm[e] * fm[e];
        lin = emb1[(size_t)lane * VOCAB + (size_t)idx];

        _Float16* hp = h0 + (size_t)b * K0PAD + lane * 16;
        v8h p0, p1;
#pragma unroll
        for (int j = 0; j < 8; j++) { p0[j] = (_Float16)fm[j]; p1[j] = (_Float16)fm[8 + j]; }
        *(v8h*)(hp)     = p0;
        *(v8h*)(hp + 8) = p1;
    }
    if (lane < NDENSE) {
        float x = xd[b * NDENSE + lane];
        lin += x * w_dense[lane];
        h0[(size_t)b * K0PAD + NFEAT * EMB + lane] = (_Float16)x;
    }
    if (lane < (K0PAD - DNN_IN)) {              // 19 pad columns
        h0[(size_t)b * K0PAD + DNN_IN + lane] = (_Float16)0.f;
    }

    // cross-lane reductions (wave32)
#pragma unroll
    for (int e = 0; e < 16; e++) {
#pragma unroll
        for (int off = 16; off; off >>= 1) fm[e] += __shfl_xor(fm[e], off, 32);
    }
#pragma unroll
    for (int off = 16; off; off >>= 1) {
        lin   += __shfl_xor(lin,   off, 32);
        sumsq += __shfl_xor(sumsq, off, 32);
    }
    float ss = 0.f;
#pragma unroll
    for (int e = 0; e < 16; e++) ss += fm[e] * fm[e];

    if (lane == 0) logit[b] = lin + 0.5f * (ss - sumsq);
}

// ---------------------------------------------------------------------------
// Zero the 16 pad columns (400..415) of a [B x 416] f16 activation buffer.
// ---------------------------------------------------------------------------
__global__ __launch_bounds__(256)
void zero_pad_kernel(_Float16* __restrict__ h)
{
    int tid = blockIdx.x * 256 + threadIdx.x;
    if (tid >= BATCH * 16) return;
    int b = tid >> 4;
    int c = tid & 15;
    h[(size_t)b * K1PAD + HIDDEN + c] = (_Float16)0.f;
}

// ---------------------------------------------------------------------------
// WMMA GEMM + bias + ReLU:  Out[B x 400] = relu(A[B x Kp] * W[K x 400] + bias)
// A: f16 row-major, stride Kp (multiple of 32). W pre-packed (pack_w_kernel).
// Block = 256 threads = 8 waves; block tile = 256 rows x 80 cols;
// wave tile = 32 rows x 80 cols (2 row-subtiles x 5 col-subtiles = 10 wmma acc).
// Each B fragment is reused for 2 wmmas -> 10 wmma per 14 b128 loads per k-step.
// grid = (B/256, 400/80) = (64, 5)
// ---------------------------------------------------------------------------
__global__ __launch_bounds__(256)
void gemm_relu_kernel(const _Float16* __restrict__ A, int Kp, int KT,
                      const _Float16* __restrict__ Wp,
                      const float* __restrict__ bias,
                      _Float16* __restrict__ Out, int ostride)
{
    const int lane = threadIdx.x & 31;
    const int wave = threadIdx.x >> 5;
    const int m0   = blockIdx.x * 256 + wave * 32;
    const int n0t  = blockIdx.y * 5;               // base n-tile (16 cols each)
    const int sel  = (lane >> 4) << 3;             // A k-offset select: 0 or 8
    const _Float16* arow0 = A + (size_t)(m0 + (lane & 15)) * Kp;
    const _Float16* arow1 = arow0 + (size_t)16 * Kp;

    v8f acc[2][5];
#pragma unroll
    for (int r = 0; r < 2; r++)
#pragma unroll
        for (int t = 0; t < 5; t++)
#pragma unroll
            for (int j = 0; j < 8; j++) acc[r][t][j] = 0.f;

    for (int kt = 0; kt < KT; ++kt) {
        const int k0 = kt * 32;
        // A fragments (16x32 f16): lane l -> row (l&15);
        // elems 0..7 -> K = k0+sel+0..7 ; elems 8..15 -> K = k0+16+sel+0..7
        v8h a0lo = *(const v8h*)(arow0 + k0 + sel);
        v8h a0hi = *(const v8h*)(arow0 + k0 + 16 + sel);
        v8h a1lo = *(const v8h*)(arow1 + k0 + sel);
        v8h a1hi = *(const v8h*)(arow1 + k0 + 16 + sel);
        // B fragments: contiguous 32B per lane from packed buffer
        v16h bf[5];
#pragma unroll
        for (int t = 0; t < 5; t++) {
            const _Float16* wp =
                Wp + (((size_t)kt * NTILES + (n0t + t)) * 32 + lane) * 16;
            v8h blo = *(const v8h*)(wp);
            v8h bhi = *(const v8h*)(wp + 8);
#pragma unroll
            for (int i = 0; i < 8; i++) { bf[t][i] = blo[i]; bf[t][8 + i] = bhi[i]; }
        }
        v16h a0, a1;
#pragma unroll
        for (int i = 0; i < 8; i++) {
            a0[i] = a0lo[i]; a0[8 + i] = a0hi[i];
            a1[i] = a1lo[i]; a1[8 + i] = a1hi[i];
        }
#pragma unroll
        for (int t = 0; t < 5; t++) {
            acc[0][t] = __builtin_amdgcn_wmma_f32_16x16x32_f16(
                false, a0, false, bf[t], (short)0, acc[0][t], false, false);
            acc[1][t] = __builtin_amdgcn_wmma_f32_16x16x32_f16(
                false, a1, false, bf[t], (short)0, acc[1][t], false, false);
        }
    }

    // Epilogue: C/D 16x16 f32 layout: VGPR j -> M = j + 8*(lane>=16); N = lane&15
#pragma unroll
    for (int r = 0; r < 2; r++) {
        const int mbase = m0 + r * 16 + sel;
#pragma unroll
        for (int t = 0; t < 5; t++) {
            const int n  = (n0t + t) * 16 + (lane & 15);
            const float bv = bias[n];
#pragma unroll
            for (int j = 0; j < 8; j++) {
                float v = acc[r][t][j] + bv;
                v = v > 0.f ? v : 0.f;
                Out[(size_t)(mbase + j) * ostride + n] = (_Float16)v;
            }
        }
    }
}

// ---------------------------------------------------------------------------
// Final: out[b] = sigmoid(logit[b] + dot(h3[b], final_w) + bias0)
// One wave per row.
// ---------------------------------------------------------------------------
__global__ __launch_bounds__(256)
void final_kernel(const _Float16* __restrict__ h3, const float* __restrict__ fw,
                  const float* __restrict__ logit, const float* __restrict__ bias,
                  float* __restrict__ out)
{
    const int lane = threadIdx.x & 31;
    const int b    = blockIdx.x * 8 + (threadIdx.x >> 5);
    float acc = 0.f;
    for (int j = lane; j < HIDDEN; j += 32)
        acc += (float)h3[(size_t)b * HIDDEN + j] * fw[j];
#pragma unroll
    for (int off = 16; off; off >>= 1) acc += __shfl_xor(acc, off, 32);
    if (lane == 0) {
        float z = acc + logit[b] + bias[0];
        out[b] = 1.f / (1.f + __expf(-z));
    }
}

// ---------------------------------------------------------------------------
extern "C" void kernel_launch(void* const* d_in, const int* in_sizes, int n_in,
                              void* d_out, int out_size, void* d_ws, size_t ws_size,
                              hipStream_t stream)
{
    const int*   xs      = (const int*)  d_in[0];   // x_sparse [B,F] i32
    const float* xd      = (const float*)d_in[1];   // x_dense  [B,D]
    const float* emb1    = (const float*)d_in[2];   // [F,V,1]
    const float* emb2    = (const float*)d_in[3];   // [F,V,E]
    const float* w_dense = (const float*)d_in[4];   // [D,1]
    const float* bias    = (const float*)d_in[5];   // [1]
    const float* w0      = (const float*)d_in[6];   // [429,400]
    const float* b0      = (const float*)d_in[7];
    const float* w1      = (const float*)d_in[8];   // [400,400]
    const float* b1      = (const float*)d_in[9];
    const float* w2      = (const float*)d_in[10];  // [400,400]
    const float* b2      = (const float*)d_in[11];
    const float* fw      = (const float*)d_in[12];  // [400,1]
    float* out = (float*)d_out;

    // workspace carve-up (aliased double-buffering: h0/h2 in bufA, h1/h3 in bufB)
    char* ws = (char*)d_ws;
    size_t off = 0;
    auto carve = [&](size_t bytes) -> void* {
        void* p = ws + off;
        off = (off + bytes + 255) & ~(size_t)255;
        return p;
    };
    _Float16* bufA  = (_Float16*)carve((size_t)BATCH * K0PAD * 2); // h0 then h2
    _Float16* bufB  = (_Float16*)carve((size_t)BATCH * K1PAD * 2); // h1 then h3
    float*    logit = (float*)   carve((size_t)BATCH * 4);
    _Float16* w0p   = (_Float16*)carve((size_t)14 * NTILES * 512 * 2);
    _Float16* w1p   = (_Float16*)carve((size_t)13 * NTILES * 512 * 2);
    _Float16* w2p   = (_Float16*)carve((size_t)13 * NTILES * 512 * 2);

    // 1) pack weights to f16 WMMA fragment order (tiny, L2-resident)
    {
        int t0 = 14 * NTILES * 512, t1 = 13 * NTILES * 512;
        pack_w_kernel<<<(t0 + 255) / 256, 256, 0, stream>>>(w0, w0p, DNN_IN, 14, HIDDEN);
        pack_w_kernel<<<(t1 + 255) / 256, 256, 0, stream>>>(w1, w1p, HIDDEN, 13, HIDDEN);
        pack_w_kernel<<<(t1 + 255) / 256, 256, 0, stream>>>(w2, w2p, HIDDEN, 13, HIDDEN);
    }

    // 2) fused embedding gather + FM + DNN-input build (writes h0 in bufA)
    fm_embed_kernel<<<BATCH / 8, 256, 0, stream>>>(xs, xd, emb1, emb2, w_dense, bufA, logit);

    // 3) zero pad columns of h1 (bufB) before layer1 consumes them
    zero_pad_kernel<<<(BATCH * 16 + 255) / 256, 256, 0, stream>>>(bufB);

    // 4) layer0: h0[B x 448] -> h1[B x 416]
    gemm_relu_kernel<<<dim3(BATCH / 256, 5), 256, 0, stream>>>(
        bufA, K0PAD, 14, w0p, b0, bufB, K1PAD);

    // 5) zero pad columns of h2 (bufA) — runs after layer0 finished reading h0
    zero_pad_kernel<<<(BATCH * 16 + 255) / 256, 256, 0, stream>>>(bufA);

    // 6) layer1: h1[B x 416] -> h2[B x 416]
    gemm_relu_kernel<<<dim3(BATCH / 256, 5), 256, 0, stream>>>(
        bufB, K1PAD, 13, w1p, b1, bufA, K1PAD);

    // 7) layer2: h2[B x 416] -> h3[B x 400] (no pad needed, stride 400)
    gemm_relu_kernel<<<dim3(BATCH / 256, 5), 256, 0, stream>>>(
        bufA, K1PAD, 13, w2p, b2, bufB, HIDDEN);

    // 8) final dot + partial logit + sigmoid
    final_kernel<<<BATCH / 8, 256, 0, stream>>>(bufB, fw, logit, bias, out);
}
